// NAFBlockDDF_11862699672009
// MI455X (gfx1250) — matrix operand order repaired and checked by reading, not compiled, not run on hardware
//
#include <hip/hip_runtime.h>
#include <hip/hip_bf16.h>
#include <stdint.h>

// ---------------------------------------------------------------------------
// NAFBlock + DDF fused pipeline for gfx1250 (wave32, WMMA 16x16x32 f16)
// B=4, c=32, dw=64, H=W=256, k=3, weight grid 32x32 bilinearly upsampled.
// ---------------------------------------------------------------------------

#define BB   4
#define CC   32
#define DW   64
#define HH   256
#define WW_  256
#define HW   (HH * WW_)          // 65536
#define EPS  1e-6f

typedef __attribute__((ext_vector_type(16))) _Float16 v16h;
typedef __attribute__((ext_vector_type(8)))  float    v8f;
typedef __attribute__((ext_vector_type(4)))  int      i32x4;

union Frag16 { v16h h; unsigned u[8]; };
union PackH2 { _Float16 h[2]; unsigned u; };
union H16    { uint4 v[2]; _Float16 h[16]; };

// ---- optional CDNA5 async global->LDS DMA path (ASYNCcnt) -----------------
#if defined(__has_builtin)
#if __has_builtin(__builtin_amdgcn_global_load_async_to_lds_b128) && \
    __has_builtin(__builtin_amdgcn_global_load_async_to_lds_b32)  && \
    __has_builtin(__builtin_amdgcn_s_wait_asynccnt)
#define HAVE_ASYNC_LDS 1
#endif
#endif

#define AS1_ __attribute__((address_space(1)))
#define AS3_ __attribute__((address_space(3)))

// Build B fragment (32x16, f16) for V_WMMA_F32_16X16X32_F16.
// Weight table layout: row-major (O, 32) f16 -> element (K,N) at w[N*32+K].
// Lanes 0-15: N=lane, K 0..7,16..23. Lanes 16-31: N=lane-16, K 8..15,24..31.
__device__ __forceinline__ v16h load_bfrag(const _Float16* wtab, int nBase, int tid) {
    const int n  = tid & 15;
    const int lh = tid >> 4;
    const unsigned* row = (const unsigned*)(wtab + (size_t)(nBase + n) * 32 + lh * 8);
    Frag16 B;
#pragma unroll
    for (int v = 0; v < 8; ++v) B.u[v] = row[(v < 4) ? v : v + 4];
    return B.h;
}

// Build A fragment (16x32, f16) from an LDS tile laid out [pixel][channel].
__device__ __forceinline__ v16h load_afrag(const _Float16* tile, int tid) {
    const int m  = tid & 15;
    const int lh = tid >> 4;
    const unsigned* arow = (const unsigned*)(tile + m * 32);
    Frag16 A;
#pragma unroll
    for (int v = 0; v < 8; ++v) {
        int kb = ((v < 4) ? 2 * v : 16 + 2 * (v - 4)) + lh * 8;   // even
        A.u[v] = arow[kb >> 1];
    }
    return A.h;
}

__device__ __forceinline__ v8f wmma_f16(v16h a, v16h b, v8f c) {
    return __builtin_amdgcn_wmma_f32_16x16x32_f16(false, a, false, b, (short)0, c,
                                                  false, false);
}

// ---------------------------------------------------------------------------
// Kernel 0: convert 1x1-conv weights to f16 table; zero pooling accumulators.
// wf16 layout: conv1 [0,2048) conv3 [2048,3072) conv4 [3072,5120) conv5 [5120,6144)
// ---------------------------------------------------------------------------
__global__ void k_prep(const float* __restrict__ w1, const float* __restrict__ w3,
                       const float* __restrict__ w4, const float* __restrict__ w5,
                       _Float16* __restrict__ wf, float* __restrict__ sums) {
    int t = threadIdx.x;
    for (int i = t; i < 2048; i += 256) wf[i]        = (_Float16)w1[i];
    for (int i = t; i < 1024; i += 256) wf[2048 + i] = (_Float16)w3[i];
    for (int i = t; i < 2048; i += 256) wf[3072 + i] = (_Float16)w4[i];
    for (int i = t; i < 1024; i += 256) wf[5120 + i] = (_Float16)w5[i];
    if (t < BB * CC) sums[t] = 0.0f;
}

// ---------------------------------------------------------------------------
// Kernel A: LayerNorm2d(norm1) + conv1 (32->64) via WMMA. x1 stored f16.
// One wave = 16 consecutive pixels. Block = 8 waves.
// ---------------------------------------------------------------------------
__global__ __launch_bounds__(256) void k_ln1_conv1(
        const float* __restrict__ inp, const _Float16* __restrict__ wf,
        const float* __restrict__ b1,  const float* __restrict__ n1w,
        const float* __restrict__ n1b, _Float16* __restrict__ x1) {
    __shared__ float    s_f32[8][CC * 16];      // [ch][pix]
    __shared__ _Float16 s_a[8][16 * CC];        // [pix][ch]

    const int tid  = threadIdx.x & 31;
    const int wave = threadIdx.x >> 5;
    const long gp0 = ((long)blockIdx.x * 8 + wave) * 16;
    const int  b   = (int)(gp0 >> 16);
    const int  hw0 = (int)(gp0 & (HW - 1));

    // stage: lane == channel, 16 contiguous pixels (64B) -> LDS [ch][pix]
    {
        const int c = tid;
        const float* src = inp + ((size_t)b * CC + c) * HW + hw0;
        float* dst = &s_f32[wave][c * 16];
#ifdef HAVE_ASYNC_LDS
#pragma unroll
        for (int i = 0; i < 4; ++i)
            __builtin_amdgcn_global_load_async_to_lds_b128(
                (AS1_ i32x4*)(src + 4 * i), (AS3_ i32x4*)(dst + 4 * i), 0, 0);
        __builtin_amdgcn_s_wait_asynccnt(0);
#else
#pragma unroll
        for (int i = 0; i < 4; ++i) {
            float4 q = ((const float4*)src)[i];
            dst[4 * i + 0] = q.x;
            dst[4 * i + 1] = q.y;
            dst[4 * i + 2] = q.z;
            dst[4 * i + 3] = q.w;
        }
#endif
    }
    __syncthreads();

    // LayerNorm over channels per pixel (both lane-halves compute stats)
    {
        const int p = tid & 15;
        float mu = 0.f;
#pragma unroll
        for (int c = 0; c < CC; ++c) mu += s_f32[wave][c * 16 + p];
        mu *= (1.0f / CC);
        float var = 0.f;
#pragma unroll
        for (int c = 0; c < CC; ++c) {
            float d = s_f32[wave][c * 16 + p] - mu;
            var += d * d;
        }
        var *= (1.0f / CC);
        const float inv = rsqrtf(var + EPS);
        const int c0 = (tid < 16) ? 0 : 16;
#pragma unroll
        for (int i = 0; i < 16; ++i) {
            int c = c0 + i;
            float v = (s_f32[wave][c * 16 + p] - mu) * inv * n1w[c] + n1b[c];
            s_a[wave][p * CC + c] = (_Float16)v;
        }
    }
    __syncthreads();

    const v16h A = load_afrag(s_a[wave], tid);
    const int  n  = tid & 15;
    const int  lh = tid >> 4;
    v8f zero = {};
#pragma unroll
    for (int t = 0; t < 4; ++t) {
        v16h Bf = load_bfrag(wf, t * 16, tid);
        v8f  d  = wmma_f16(A, Bf, zero);
        const float bias = b1[t * 16 + n];
        _Float16* dst = x1 + ((size_t)b * DW + t * 16 + n) * HW + hw0 + lh * 8;
        union { unsigned u[4]; uint4 v; } pk;
#pragma unroll
        for (int r2 = 0; r2 < 4; ++r2) {
            PackH2 pp;
            pp.h[0] = (_Float16)(d[2 * r2 + 0] + bias);
            pp.h[1] = (_Float16)(d[2 * r2 + 1] + bias);
            pk.u[r2] = pp.u;
        }
        *(uint4*)dst = pk.v;
    }
}

// ---------------------------------------------------------------------------
// Kernel B: bilinear-upsampled dynamic 3x3 per-channel conv + SimpleGate,
// plus block-reduced global-average-pool accumulation (atomicAdd).
// Block = one (b, cc, h) row, thread = pixel w; processes channels cc & cc+32.
// Only the two needed source rows (h0,h1) of each 32x32 weight map are staged
// to LDS once per block: 2ch x 9 taps x 2 rows x 32 = 1152 floats (4.6 KB).
// ---------------------------------------------------------------------------
__global__ __launch_bounds__(256) void k_ddf_gate(
        const _Float16* __restrict__ x1, const float* __restrict__ w2,
        _Float16* __restrict__ g, float* __restrict__ sums) {
    __shared__ float lw[2 * 9 * 2 * 32];     // [s][kidx][row][32]
    __shared__ float red[256];

    const int idx = blockIdx.x;
    const int h   = idx & 255;
    const int cc  = (idx >> 8) & 31;
    const int b   = idx >> 13;
    const int w   = threadIdx.x;

    // half-pixel-center bilinear coordinates, 256 -> 32 (scale 1/8)
    const float sh = (h + 0.5f) * 0.125f - 0.5f;       // uniform per block
    const float fh = floorf(sh);
    const float ah = sh - fh;
    const int h0 = min(max((int)fh, 0), 31), h1 = min(max((int)fh + 1, 0), 31);

    float sw = (w + 0.5f) * 0.125f - 0.5f;             // per thread
    float fw = floorf(sw);
    float aw = sw - fw;
    int w0 = min(max((int)fw, 0), 31), w1 = min(max((int)fw + 1, 0), 31);

    // stage the two weight rows for all 9 taps and both channels
    for (int i = threadIdx.x; i < 1152; i += 256) {
        const int x    = i & 31;
        const int row  = (i >> 5) & 1;
        const int t2   = i >> 6;            // 0..17
        const int kidx = t2 % 9;
        const int s    = t2 / 9;
        const int ch   = cc + s * CC;
        const float* src = w2 + (((size_t)b * DW + ch) * 9 + kidx) * 1024
                              + (row ? h1 : h0) * 32 + x;
#ifdef HAVE_ASYNC_LDS
        __builtin_amdgcn_global_load_async_to_lds_b32(
            (AS1_ int*)src, (AS3_ int*)&lw[i], 0, 0);
#else
        lw[i] = *src;
#endif
    }
#ifdef HAVE_ASYNC_LDS
    __builtin_amdgcn_s_wait_asynccnt(0);
#endif
    __syncthreads();

    float acc2[2];
#pragma unroll
    for (int s = 0; s < 2; ++s) {
        const int ch = cc + s * CC;
        const _Float16* xc = x1 + ((size_t)b * DW + ch) * HW;
        float a = 0.f;
#pragma unroll
        for (int ky = 0; ky < 3; ++ky) {
#pragma unroll
            for (int kx = 0; kx < 3; ++kx) {
                const float* r0 = &lw[((s * 9 + ky * 3 + kx) * 2 + 0) * 32];
                const float* r1 = r0 + 32;
                const float top = r0[w0] + (r0[w1] - r0[w0]) * aw;
                const float bot = r1[w0] + (r1[w1] - r1[w0]) * aw;
                const float wgt = top + (bot - top) * ah;
                const int yy = h + ky - 1, xx = w + kx - 1;
                float xv = 0.f;
                if ((unsigned)yy < 256u && (unsigned)xx < 256u)
                    xv = (float)xc[yy * WW_ + xx];
                a += wgt * xv;
            }
        }
        acc2[s] = a;
    }
    const float gv = acc2[0] * acc2[1];
    g[((size_t)b * CC + cc) * HW + h * WW_ + w] = (_Float16)gv;

    red[threadIdx.x] = gv;
    __syncthreads();
    for (int off = 128; off > 0; off >>= 1) {
        if (threadIdx.x < off) red[threadIdx.x] += red[threadIdx.x + off];
        __syncthreads();
    }
    if (threadIdx.x == 0) atomicAdd(&sums[b * CC + cc], red[0]);
}

// ---------------------------------------------------------------------------
// Kernel C: sca 32x32 GEMV on pooled means: s[b][o] = sca_b[o] + W.mean
// ---------------------------------------------------------------------------
__global__ void k_sca(const float* __restrict__ sums, const float* __restrict__ sw,
                      const float* __restrict__ sb, float* __restrict__ sbuf) {
    int t = threadIdx.x;
    if (t < BB * CC) {
        int b = t >> 5, o = t & 31;
        float acc = sb[o];
        const float inv = 1.0f / (float)HW;
#pragma unroll
        for (int c = 0; c < CC; ++c)
            acc += sw[o * CC + c] * sums[b * CC + c] * inv;
        sbuf[t] = acc;
    }
}

// ---------------------------------------------------------------------------
// Kernel D: fused tail. x = g*s -> conv3(WMMA) -> y = inp + x*beta
//           -> LN(norm2) -> conv4(WMMA) -> gate -> conv5(WMMA)
//           -> out = y + x*gamma
// One wave = 16 pixels, block = 8 waves.
// ---------------------------------------------------------------------------
__global__ __launch_bounds__(256) void k_tail(
        const float* __restrict__ inp, const _Float16* __restrict__ g,
        const float* __restrict__ sbuf, const _Float16* __restrict__ wf,
        const float* __restrict__ b3, const float* __restrict__ b4,
        const float* __restrict__ b5, const float* __restrict__ n2w,
        const float* __restrict__ n2b, const float* __restrict__ beta,
        const float* __restrict__ gamma, float* __restrict__ out) {
    __shared__ float    s_y[8][16 * CC];     // [pix][ch] f32
    __shared__ _Float16 s_a[8][16 * CC];     // [pix][ch] f16 (reused per phase)

    const _Float16* wf3 = wf + 2048;
    const _Float16* wf4 = wf + 3072;
    const _Float16* wf5 = wf + 5120;

    const int tid  = threadIdx.x & 31;
    const int wave = threadIdx.x >> 5;
    const long gp0 = ((long)blockIdx.x * 8 + wave) * 16;
    const int  b   = (int)(gp0 >> 16);
    const int  hw0 = (int)(gp0 & (HW - 1));
    const int  n   = tid & 15;
    const int  lh  = tid >> 4;
    v8f zero = {};

    // ---- stage x = g * s as f16 A-tile ----
    {
        const int c = tid;
        const float sval = sbuf[b * CC + c];
        const _Float16* gsrc = g + ((size_t)b * CC + c) * HW + hw0;
        H16 gv;
        gv.v[0] = ((const uint4*)gsrc)[0];
        gv.v[1] = ((const uint4*)gsrc)[1];
#pragma unroll
        for (int p = 0; p < 16; ++p)
            s_a[wave][p * CC + c] = (_Float16)((float)gv.h[p] * sval);
    }
    __syncthreads();

    // ---- conv3 (32 -> 32), residual y = inp + x*beta ----
    float y[2][8];
    {
        const v16h A = load_afrag(s_a[wave], tid);
#pragma unroll
        for (int t = 0; t < 2; ++t) {
            v16h Bf = load_bfrag(wf3, t * 16, tid);
            v8f  d  = wmma_f16(A, Bf, zero);
            const int ch = t * 16 + n;
            const float bia = b3[ch], bet = beta[ch];
            const float* isrc = inp + ((size_t)b * CC + ch) * HW + hw0 + lh * 8;
#pragma unroll
            for (int r = 0; r < 8; ++r) {
                y[t][r] = isrc[r] + (d[r] + bia) * bet;
                s_y[wave][(lh * 8 + r) * CC + ch] = y[t][r];
            }
        }
    }
    __syncthreads();

    // ---- LayerNorm(norm2) on y -> f16 A-tile ----
    {
        const int p = tid & 15;
        float mu = 0.f;
#pragma unroll
        for (int c = 0; c < CC; ++c) mu += s_y[wave][p * CC + c];
        mu *= (1.0f / CC);
        float var = 0.f;
#pragma unroll
        for (int c = 0; c < CC; ++c) {
            float d = s_y[wave][p * CC + c] - mu;
            var += d * d;
        }
        var *= (1.0f / CC);
        const float inv = rsqrtf(var + EPS);
        const int c0 = (tid < 16) ? 0 : 16;
#pragma unroll
        for (int i = 0; i < 16; ++i) {
            int c = c0 + i;
            float v = (s_y[wave][p * CC + c] - mu) * inv * n2w[c] + n2b[c];
            s_a[wave][p * CC + c] = (_Float16)v;
        }
    }
    __syncthreads();

    // ---- conv4 (32 -> 64) + SimpleGate -> f16 A-tile ----
    {
        const v16h A = load_afrag(s_a[wave], tid);
        v8f d4[4];
#pragma unroll
        for (int t = 0; t < 4; ++t) {
            v16h Bf = load_bfrag(wf4, t * 16, tid);
            d4[t] = wmma_f16(A, Bf, zero);
        }
        __syncthreads();   // everyone done reading s_a before gate overwrite
#pragma unroll
        for (int t = 0; t < 2; ++t) {
            const int ch = t * 16 + n;
            const float ba = b4[ch], bb = b4[CC + ch];
#pragma unroll
            for (int r = 0; r < 8; ++r) {
                float v1 = d4[t][r] + ba;
                float v2 = d4[t + 2][r] + bb;
                s_a[wave][(lh * 8 + r) * CC + ch] = (_Float16)(v1 * v2);
            }
        }
    }
    __syncthreads();

    // ---- conv5 (32 -> 32), out = y + x*gamma ----
    {
        const v16h A = load_afrag(s_a[wave], tid);
#pragma unroll
        for (int t = 0; t < 2; ++t) {
            v16h Bf = load_bfrag(wf5, t * 16, tid);
            v8f  d  = wmma_f16(A, Bf, zero);
            const int ch = t * 16 + n;
            const float bia = b5[ch], gam = gamma[ch];
            float* op = out + ((size_t)b * CC + ch) * HW + hw0 + lh * 8;
            float4 o0, o1;
            o0.x = y[t][0] + (d[0] + bia) * gam;
            o0.y = y[t][1] + (d[1] + bia) * gam;
            o0.z = y[t][2] + (d[2] + bia) * gam;
            o0.w = y[t][3] + (d[3] + bia) * gam;
            o1.x = y[t][4] + (d[4] + bia) * gam;
            o1.y = y[t][5] + (d[5] + bia) * gam;
            o1.z = y[t][6] + (d[6] + bia) * gam;
            o1.w = y[t][7] + (d[7] + bia) * gam;
            ((float4*)op)[0] = o0;
            ((float4*)op)[1] = o1;
        }
    }
}

// ---------------------------------------------------------------------------
extern "C" void kernel_launch(void* const* d_in, const int* in_sizes, int n_in,
                              void* d_out, int out_size, void* d_ws, size_t ws_size,
                              hipStream_t stream) {
    const float* inp    = (const float*)d_in[0];
    const float* w2     = (const float*)d_in[1];
    const float* conv1w = (const float*)d_in[2];
    const float* conv1b = (const float*)d_in[3];
    const float* conv3w = (const float*)d_in[4];
    const float* conv3b = (const float*)d_in[5];
    const float* scaw   = (const float*)d_in[6];
    const float* scab   = (const float*)d_in[7];
    const float* conv4w = (const float*)d_in[8];
    const float* conv4b = (const float*)d_in[9];
    const float* conv5w = (const float*)d_in[10];
    const float* conv5b = (const float*)d_in[11];
    const float* n1w    = (const float*)d_in[12];
    const float* n1b    = (const float*)d_in[13];
    const float* n2w    = (const float*)d_in[14];
    const float* n2b    = (const float*)d_in[15];
    const float* beta   = (const float*)d_in[16];
    const float* gamma  = (const float*)d_in[17];
    float* out = (float*)d_out;

    // workspace carve-up
    char* ws = (char*)d_ws;
    _Float16* wf16  = (_Float16*)(ws);                         // 12 KB (pad 16 KB)
    _Float16* x1buf = (_Float16*)(ws + (16 << 10));            // 32 MB
    _Float16* gbuf  = (_Float16*)(ws + (16 << 10) + ((size_t)32 << 20)); // 16 MB
    float*    sums  = (float*)(ws + (16 << 10) + ((size_t)48 << 20));
    float*    sbuf  = sums + 256;

    const int tiles   = BB * HW / 16;         // 16384 waves
    const int blocksA = tiles / 8;            // 2048

    k_prep<<<1, 256, 0, stream>>>(conv1w, conv3w, conv4w, conv5w, wf16, sums);
    k_ln1_conv1<<<blocksA, 256, 0, stream>>>(inp, wf16, conv1b, n1w, n1b, x1buf);
    k_ddf_gate<<<BB * CC * HH, 256, 0, stream>>>(x1buf, w2, gbuf, sums);
    k_sca<<<1, 128, 0, stream>>>(sums, scaw, scab, sbuf);
    k_tail<<<blocksA, 256, 0, stream>>>(inp, gbuf, sbuf, wf16, conv3b, conv4b,
                                        conv5b, n2w, n2b, beta, gamma, out);
}
